// GRU_38302518346004
// MI455X (gfx1250) — matrix operand order repaired and checked by reading, not compiled, not run on hardware
//
#include <hip/hip_runtime.h>
#include <math.h>

// ---------------- types for WMMA ----------------
typedef __attribute__((ext_vector_type(16))) __bf16 v16bf;
typedef __attribute__((ext_vector_type(8)))  float  v8f;

union FragBF16 {
    v16bf bf;
    uint4 u4[2];
};

// ---------------- f32 -> bf16 (RNE) elementwise convert ----------------
__global__ __launch_bounds__(256) void f32_to_bf16_kernel(
    const float* __restrict__ in, unsigned short* __restrict__ out, int n)
{
    int i = blockIdx.x * blockDim.x + threadIdx.x;
    if (i < n) {
        unsigned int b = __float_as_uint(in[i]);
        unsigned int r = (b + 0x7FFFu + ((b >> 16) & 1u)) >> 16;
        out[i] = (unsigned short)r;
    }
}

// ---------------- NT GEMM: C[M,N] = A[M,K](bf16) * B[N,K](bf16)^T, f32 acc ----
// block = 128 threads (4 waves). Block tile: 32 (M) x 256 (N).
// Each wave: 32 x 64 via 2x4 grid of 16x16 WMMA tiles, K-step 32.
__global__ __launch_bounds__(128) void gemm_bf16_nt_kernel(
    const unsigned short* __restrict__ A,   // [M,K]
    const unsigned short* __restrict__ B,   // [N,K]
    float* __restrict__ C,                  // [M,N]
    int M, int N, int K)
{
    const int lane   = threadIdx.x & 31;
    const int wave   = threadIdx.x >> 5;          // 0..3
    const int lane16 = lane & 15;
    const int hi     = lane >> 4;                 // 0 or 1

    const int mBase = blockIdx.y * 32;
    const int nBase = blockIdx.x * 256 + wave * 64;

    // Per-lane K offsets per documented fragment layouts
    const int aK = hi * 8;    // A: lanes 0-15 -> K 0..7 & 16..23 ; lanes 16-31 -> K 8..15 & 24..31
    const int bK = hi * 8;    // B: lanes 0-15 -> K 0..7 & 16..23 ; lanes 16-31 -> K 8..15 & 24..31

    const unsigned short* aRow[2];
    aRow[0] = A + (size_t)(mBase + lane16) * K + aK;
    aRow[1] = aRow[0] + (size_t)16 * K;

    const unsigned short* bCol[4];
#pragma unroll
    for (int nt = 0; nt < 4; ++nt)
        bCol[nt] = B + (size_t)(nBase + nt * 16 + lane16) * K + bK;

    v8f acc[2][4] = {};

    for (int k = 0; k < K; k += 32) {
        if (k + 64 <= K) {
            __builtin_prefetch(aRow[0] + k + 32, 0, 1);
            __builtin_prefetch(bCol[0] + k + 32, 0, 1);
            __builtin_prefetch(bCol[2] + k + 32, 0, 1);
        }
        FragBF16 a[2], b[4];
#pragma unroll
        for (int mt = 0; mt < 2; ++mt) {
            a[mt].u4[0] = *(const uint4*)(aRow[mt] + k);
            a[mt].u4[1] = *(const uint4*)(aRow[mt] + k + 16);
        }
#pragma unroll
        for (int nt = 0; nt < 4; ++nt) {
            b[nt].u4[0] = *(const uint4*)(bCol[nt] + k);
            b[nt].u4[1] = *(const uint4*)(bCol[nt] + k + 16);
        }
#pragma unroll
        for (int mt = 0; mt < 2; ++mt)
#pragma unroll
            for (int nt = 0; nt < 4; ++nt)
                acc[mt][nt] = __builtin_amdgcn_wmma_f32_16x16x32_bf16(
                    false, a[mt].bf, false, b[nt].bf,
                    (short)0, acc[mt][nt], false, false);
    }

    // Store: VGPR r -> M = mt*16 + r + hi*8 ; N = nBase + nt*16 + lane16
#pragma unroll
    for (int mt = 0; mt < 2; ++mt) {
#pragma unroll
        for (int nt = 0; nt < 4; ++nt) {
            int n = nBase + nt * 16 + lane16;
            float* cp = C + (size_t)(mBase + mt * 16 + hi * 8) * N + n;
#pragma unroll
            for (int r = 0; r < 8; ++r)
                cp[(size_t)r * N] = acc[mt][nt][r];
        }
    }
}

// ---------------- GRU recurrent scan ----------------
// One 64-thread workgroup per (b, head). Weights cached in LDS (48 KB / WGP 320 KB).
__global__ __launch_bounds__(64) void gru_scan_kernel(
    const float* __restrict__ proj,  // [B, S, 6144] = [i | f | r]
    const float* __restrict__ sw,    // [96, 64, 64]  (W | Wf | Wr)
    float* __restrict__ y,           // [B, S, 2048]
    int S)
{
    const int H = 32, D = 64;
    const int b = blockIdx.x / H;
    const int head = blockIdx.x % H;
    const int e = threadIdx.x;

    __shared__ float Wi[64 * 64];
    __shared__ float Wf[64 * 64];
    __shared__ float Wr[64 * 64];
    __shared__ float h[64];
    __shared__ float rh[64];

    for (int idx = e; idx < D * D; idx += 64) {
        Wi[idx] = sw[(size_t)(0 * H + head) * D * D + idx];
        Wf[idx] = sw[(size_t)(1 * H + head) * D * D + idx];
        Wr[idx] = sw[(size_t)(2 * H + head) * D * D + idx];
    }
    h[e] = 0.0f;
    __syncthreads();

    const size_t rowStride = 6144;
    const float* pBase = proj + (size_t)b * S * rowStride + head * 64 + e;
    float* yBase = y + (size_t)b * S * 2048 + head * 64 + e;

    for (int t = 0; t < S; ++t) {
        const float* p = pBase + (size_t)t * rowStride;
        float it = p[0];
        float ft = p[2048];
        float rt = p[4096];

        float dr = 0.0f, df = 0.0f;
#pragma unroll 8
        for (int d = 0; d < D; ++d) {
            float hd = h[d];
            dr += hd * Wr[d * 64 + e];
            df += hd * Wf[d * 64 + e];
        }
        float rg = 1.0f / (1.0f + __expf(-(rt + dr)));
        float fg = 1.0f / (1.0f + __expf(-(ft + df)));

        rh[e] = rg * h[e];
        __syncthreads();

        float di = 0.0f;
#pragma unroll 8
        for (int d = 0; d < D; ++d)
            di += rh[d] * Wi[d * 64 + e];
        float cand = tanhf(it + di);
        float hn = fg * h[e] + (1.0f - fg) * cand;
        __syncthreads();

        h[e] = hn;
        yBase[(size_t)t * 2048] = hn;
        __syncthreads();
    }
}

// ---------------- RMSNorm + weight + bf16 convert ----------------
__global__ __launch_bounds__(256) void rmsnorm_bf16_kernel(
    const float* __restrict__ y,       // [rows, D]
    const float* __restrict__ w,       // [D]
    unsigned short* __restrict__ out,  // [rows, D] bf16
    int D)
{
    const int row = blockIdx.x;
    const float* yr = y + (size_t)row * D;

    float s = 0.0f;
    for (int i = threadIdx.x; i < D; i += 256) {
        float v = yr[i];
        s += v * v;
    }
    __shared__ float red[256];
    red[threadIdx.x] = s;
    __syncthreads();
    for (int off = 128; off > 0; off >>= 1) {
        if (threadIdx.x < off) red[threadIdx.x] += red[threadIdx.x + off];
        __syncthreads();
    }
    float scale = rsqrtf(red[0] / (float)D + 1e-6f);

    for (int i = threadIdx.x; i < D; i += 256) {
        float v = yr[i] * scale * w[i];
        unsigned int bbits = __float_as_uint(v);
        unsigned int r = (bbits + 0x7FFFu + ((bbits >> 16) & 1u)) >> 16;
        out[(size_t)row * D + i] = (unsigned short)r;
    }
}

// ---------------- host-side launch ----------------
extern "C" void kernel_launch(void* const* d_in, const int* in_sizes, int n_in,
                              void* d_out, int out_size, void* d_ws, size_t ws_size,
                              hipStream_t stream)
{
    const float* input = (const float*)d_in[0];  // [4, 2048, 2048]
    const float* w_in  = (const float*)d_in[1];  // [6144, 2048]
    const float* sw    = (const float*)d_in[2];  // [96, 64, 64]
    const float* normw = (const float*)d_in[3];  // [2048]
    const float* w_out = (const float*)d_in[4];  // [2048, 2048]
    float* out = (float*)d_out;                  // [4, 2048, 2048]

    const int Bn = 4, S = 2048, IN = 2048, ST = 2048, OUT = 2048;
    const int M  = Bn * S;     // 8192
    const int N1 = 3 * ST;     // 6144

    // Carve workspace
    char* ws = (char*)d_ws;
    size_t off = 0;
    auto carve = [&](size_t bytes) -> void* {
        void* p = ws + off;
        off = (off + bytes + 255) & ~(size_t)255;
        return p;
    };
    unsigned short* aBf    = (unsigned short*)carve((size_t)M * IN * 2);
    unsigned short* wInBf  = (unsigned short*)carve((size_t)N1 * IN * 2);
    float*          proj   = (float*)carve((size_t)M * N1 * 4);
    float*          ybuf   = (float*)carve((size_t)M * ST * 4);
    unsigned short* yBf    = (unsigned short*)carve((size_t)M * ST * 2);
    unsigned short* wOutBf = (unsigned short*)carve((size_t)OUT * ST * 2);

    // 1) convert inputs to bf16
    {
        int n = M * IN;
        f32_to_bf16_kernel<<<(n + 255) / 256, 256, 0, stream>>>(input, aBf, n);
        n = N1 * IN;
        f32_to_bf16_kernel<<<(n + 255) / 256, 256, 0, stream>>>(w_in, wInBf, n);
        n = OUT * ST;
        f32_to_bf16_kernel<<<(n + 255) / 256, 256, 0, stream>>>(w_out, wOutBf, n);
    }

    // 2) proj = input @ w_in^T   (M=8192, N=6144, K=2048)
    {
        dim3 grid(N1 / 256, M / 32);
        gemm_bf16_nt_kernel<<<grid, 128, 0, stream>>>(aBf, wInBf, proj, M, N1, IN);
    }

    // 3) recurrent scan: 128 independent (b, head) recurrences
    gru_scan_kernel<<<Bn * 32, 64, 0, stream>>>(proj, sw, ybuf, S);

    // 4) RMSNorm + norm_weight, convert to bf16
    rmsnorm_bf16_kernel<<<M, 256, 0, stream>>>(ybuf, normw, yBf, ST);

    // 5) out = y_norm @ w_out^T  (M=8192, N=2048, K=2048)
    {
        dim3 grid(OUT / 256, M / 32);
        gemm_bf16_nt_kernel<<<grid, 128, 0, stream>>>(yBf, wOutBf, out, M, OUT, ST);
    }
}
